// DeltaNet_31877247271485
// MI455X (gfx1250) — matrix operand order repaired and checked by reading, not compiled, not run on hardware
//
#include <hip/hip_runtime.h>
#include <hip/hip_bf16.h>

typedef __bf16 bf16;
typedef __attribute__((ext_vector_type(16))) __bf16 v16bf;
typedef __attribute__((ext_vector_type(8)))  float  v8f;
typedef __attribute__((ext_vector_type(4)))  int    v4i_t;

#define B_   2
#define H_   8
#define L_   4096
#define D_   128
#define C_   32
#define NC_  (L_ / C_)     // 128 chunks per head
#define BH_  (B_ * H_)     // 16
#define NCH_ (BH_ * NC_)   // 2048 total chunks
#define DVB_ 32            // dv slice width per scan block
#define NSL_ (D_ / DVB_)   // 4 slices

// padded LDS leading dimensions (odd multiples of 8 elems -> conflict-free,
// 16B-aligned fragment groups)
#define QLD 136            // for 128-wide row-major (q, w)
#define KLD 40             // for 32-wide rows (kt, attn, u, S columns of 32)
#define SLD 136            // state shadow: 32 columns x 128 rows (+pad)

// ---------------------------------------------------------------------------
// Async global -> LDS staging (CDNA5 GLOBAL_LOAD_ASYNC_TO_LDS_B128, ASYNCcnt)
// ---------------------------------------------------------------------------
#if defined(__AMDGCN__) && \
    __has_builtin(__builtin_amdgcn_global_load_async_to_lds_b128) && \
    __has_builtin(__builtin_amdgcn_s_wait_asynccnt)
#define ASYNC_STAGE 1
#else
#define ASYNC_STAGE 0
#endif

#if ASYNC_STAGE
typedef __attribute__((address_space(1))) v4i_t* as1_v4i;
typedef __attribute__((address_space(3))) v4i_t* as3_v4i;
#endif

__device__ __forceinline__ void cp16(const bf16* g, bf16* l) {
#if ASYNC_STAGE
  __builtin_amdgcn_global_load_async_to_lds_b128((as1_v4i)g, (as3_v4i)l, 0, 0);
#else
  *(uint4*)l = *(const uint4*)g;
#endif
}

#if ASYNC_STAGE
#define STAGE_WAIT(n) __builtin_amdgcn_s_wait_asynccnt(n)
#else
#define STAGE_WAIT(n)
#endif

// ---------------------------------------------------------------------------
// bf16 bit utilities
// ---------------------------------------------------------------------------
__device__ __forceinline__ unsigned pack2(float a, float b) {
  union { bf16 h; unsigned short s; } x, y;
  x.h = (bf16)a; y.h = (bf16)b;
  return (unsigned)x.s | ((unsigned)y.s << 16);
}
__device__ __forceinline__ float bflo(unsigned w) {
  union { unsigned v; float f; } x; x.v = w << 16; return x.f;
}
__device__ __forceinline__ float bfhi(unsigned w) {
  union { unsigned v; float f; } x; x.v = w & 0xffff0000u; return x.f;
}

// ---------------------------------------------------------------------------
// WMMA helpers: CDNA5 v_wmma_f32_16x16x32_bf16, D = A(16x32) * B(32x16) + C
// ---------------------------------------------------------------------------
__device__ __forceinline__ v8f wmma_bf16(v16bf a, v16bf b, v8f c) {
  return __builtin_amdgcn_wmma_f32_16x16x32_bf16(
      false, a, false, b, (short)0, c, false, false);
}

// A fragment (16x32) from row-major [rows x ld] at (m0, k0); groups of 8
// contiguous bf16 -> ds_load_b128.
__device__ __forceinline__ v16bf load_a_frag(const bf16* src, int ld, int m0,
                                             int k0, int lane) {
  int m  = m0 + (lane & 15);
  int kh = (lane >> 4) << 3;
  const bf16* row = src + m * ld + k0;
  v16bf a;
#pragma unroll
  for (int i = 0; i < 4; ++i) {
    a[2 * i + 0] = row[kh + 2 * i + 0];
    a[2 * i + 1] = row[kh + 2 * i + 1];
    a[2 * i + 8] = row[16 + kh + 2 * i + 0];
    a[2 * i + 9] = row[16 + kh + 2 * i + 1];
  }
  return a;
}

// B fragment (32x16) from COLUMN-major storage: element (k,n) at src[n*ldc+k];
// two groups of 8 contiguous bf16 per lane (k = kh..kh+7, 16+kh..) -> 2x ds_load_b128.
__device__ __forceinline__ v16bf load_b_frag_col(const bf16* src, int ldc,
                                                 int k0, int n0, int lane) {
  int n  = n0 + (lane & 15);
  int kh = (lane >> 4) << 3;
  const bf16* p = src + n * ldc + k0 + kh;
  v16bf b;
#pragma unroll
  for (int e = 0; e < 16; ++e) b[e] = p[e < 8 ? e : e + 8];
  return b;
}

// ---------------------------------------------------------------------------
// Kernel 1: per-chunk preprocessing (2048 blocks x 128 threads)
// Outputs: Qw,Ww row-major [c x d]; Ktw,Utw TRANSPOSED [d x c]; Aw [c x c].
// ---------------------------------------------------------------------------
__global__ __launch_bounds__(128) void deltanet_prep(
    const float* __restrict__ qg, const float* __restrict__ kg,
    const float* __restrict__ vg, const float* __restrict__ betag,
    bf16* __restrict__ Qw, bf16* __restrict__ Ktw, bf16* __restrict__ Ww,
    bf16* __restrict__ Utw, bf16* __restrict__ Aw) {
  __shared__ __align__(16) float qs[C_ * D_];
  __shared__ __align__(16) float ks[C_ * D_];
  __shared__ __align__(16) float vs[C_ * D_];
  __shared__ float invm[C_ * C_];
  __shared__ float redq[128], redk[128];
  __shared__ float bet[C_], rq[C_], rk[C_], updrow[C_];

  const int tid = threadIdx.x;
  const int ch  = blockIdx.x;
  const size_t base = (size_t)ch * (C_ * D_);

#pragma unroll
  for (int r = 0; r < 8; ++r) {
    int i = tid + 128 * r;
    ((float4*)qs)[i] = ((const float4*)(qg + base))[i];
    ((float4*)ks)[i] = ((const float4*)(kg + base))[i];
    ((float4*)vs)[i] = ((const float4*)(vg + base))[i];
  }
  if (tid < C_) bet[tid] = betag[(size_t)ch * C_ + tid];
  __syncthreads();

  {
    int row = tid >> 2, p0 = (tid & 3) * 32;
    const float* qr = qs + row * D_ + p0;
    const float* kr = ks + row * D_ + p0;
    float sq = 0.f, sk = 0.f;
#pragma unroll
    for (int c = 0; c < 32; ++c) { sq += qr[c] * qr[c]; sk += kr[c] * kr[c]; }
    redq[tid] = sq; redk[tid] = sk;
  }
  __syncthreads();
  if (tid < C_) {
    float sq = redq[tid*4] + redq[tid*4+1] + redq[tid*4+2] + redq[tid*4+3];
    float sk = redk[tid*4] + redk[tid*4+1] + redk[tid*4+2] + redk[tid*4+3];
    rq[tid] = rsqrtf(sq + 1e-6f);
    rk[tid] = rsqrtf(sk + 1e-6f);
  }
  __syncthreads();

  // normalize (column owned by this thread), emit Q row-major + Kt transposed
  {
    float kloc[C_];
#pragma unroll
    for (int r = 0; r < C_; ++r) {
      int i = r * D_ + tid;
      float qv = qs[i] * rq[r];
      float kv = ks[i] * rk[r];
      qs[i] = qv; ks[i] = kv;
      Qw[base + i] = (bf16)qv;
      kloc[r] = kv;
    }
    bf16* kt = Ktw + base + (size_t)tid * C_;
#pragma unroll
    for (int p = 0; p < 4; ++p) {
      uint4 v4;
      v4.x = pack2(kloc[p*8+0], kloc[p*8+1]);
      v4.y = pack2(kloc[p*8+2], kloc[p*8+3]);
      v4.z = pack2(kloc[p*8+4], kloc[p*8+5]);
      v4.w = pack2(kloc[p*8+6], kloc[p*8+7]);
      *(uint4*)(kt + p * 8) = v4;
    }
  }
  __syncthreads();

  // attn (lower incl diag) and inv init (strict lower of -beta_i * k_i.k_j)
  for (int e = tid; e < C_ * C_; e += 128) {
    int i = e >> 5, j = e & 31;
    float dq = 0.f, dk = 0.f;
    if (j <= i) {
      const float* qi = qs + i * D_;
      const float* ki = ks + i * D_;
      const float* kj = ks + j * D_;
      for (int c = 0; c < D_; ++c) { dq += qi[c]*kj[c]; dk += ki[c]*kj[c]; }
    }
    Aw[(size_t)ch * (C_ * C_) + e] = (bf16)((j <= i) ? dq : 0.f);
    invm[e] = (j < i) ? (-bet[i] * dk) : 0.f;
  }
  __syncthreads();

  // forward substitution (row-i snapshot semantics via updrow)
  for (int i = 1; i < C_; ++i) {
    if (tid < i) {
      float acc = 0.f;
      for (int kk = 0; kk < C_; ++kk)
        acc += invm[i * C_ + kk] * invm[kk * C_ + tid];
      updrow[tid] = acc;
    }
    __syncthreads();
    if (tid < i) invm[i * C_ + tid] += updrow[tid];
    __syncthreads();
  }
  if (tid < C_) invm[tid * C_ + tid] = 1.f;
  __syncthreads();

  // u = inv @ (beta*v) -> Ut transposed ; w = inv @ (beta*k) -> W row-major
  {
    const int col = tid;
    float uloc[C_];
    for (int i = 0; i < C_; ++i) {
      float au = 0.f, aw = 0.f;
#pragma unroll
      for (int j = 0; j < C_; ++j) {
        float m = invm[i * C_ + j] * bet[j];
        au += m * vs[j * D_ + col];
        aw += m * ks[j * D_ + col];
      }
      uloc[i] = au;
      Ww[base + i * D_ + col] = (bf16)aw;
    }
    bf16* ut = Utw + base + (size_t)col * C_;
#pragma unroll
    for (int p = 0; p < 4; ++p) {
      uint4 v4;
      v4.x = pack2(uloc[p*8+0], uloc[p*8+1]);
      v4.y = pack2(uloc[p*8+2], uloc[p*8+3]);
      v4.z = pack2(uloc[p*8+4], uloc[p*8+5]);
      v4.w = pack2(uloc[p*8+6], uloc[p*8+7]);
      *(uint4*)(ut + p * 8) = v4;
    }
  }
}

// ---------------------------------------------------------------------------
// Kernel 2: sequential chunk scan with triple-buffered async staging.
// 64 blocks = 16 heads x 4 dv-slices, 128 threads = 4 waves.
// ---------------------------------------------------------------------------
__global__ __launch_bounds__(128) void deltanet_scan(
    const bf16* __restrict__ Qw, const bf16* __restrict__ Ktw,
    const bf16* __restrict__ Ww, const bf16* __restrict__ Utw,
    const bf16* __restrict__ Aw, float* __restrict__ outg,
    float* __restrict__ Sg) {
  __shared__ __align__(16) bf16 qsh[3][C_ * QLD];
  __shared__ __align__(16) bf16 wsh[3][C_ * QLD];
  __shared__ __align__(16) bf16 ktsh[3][D_ * KLD];
  __shared__ __align__(16) bf16 ash[3][C_ * KLD];
  __shared__ __align__(16) bf16 usht[3][DVB_ * KLD];
  __shared__ __align__(16) bf16 Sb[DVB_ * SLD];    // state shadow, col-major
  __shared__ __align__(16) bf16 ubt[DVB_ * KLD];   // u_s, col-major

  const int tid  = threadIdx.x;
  const int lane = tid & 31;
  const int wv   = tid >> 5;
  const int bh   = blockIdx.x >> 2;
  const int dvb  = blockIdx.x & 3;

  const int mi    = wv >> 1;
  const int ni    = wv & 1;
  const int nn    = lane & 15;
  const int mhalf = (lane >> 4) << 3;

  // 14 x cp16 per thread per chunk (wave ASYNCcnt += 14)
  auto stage = [&](int ch, int buf) {
    const size_t off  = ((size_t)bh * NC_ + ch) * (C_ * D_);
    const size_t aoff = ((size_t)bh * NC_ + ch) * (C_ * C_);
#pragma unroll
    for (int r = 0; r < 4; ++r) {
      int i = tid + 128 * r;
      int row = i >> 4, part = i & 15;
      cp16(Qw + off + (size_t)i * 8, &qsh[buf][row * QLD + part * 8]);
      cp16(Ww + off + (size_t)i * 8, &wsh[buf][row * QLD + part * 8]);
      int krow = i >> 2, kpart = i & 3;
      cp16(Ktw + off + (size_t)i * 8, &ktsh[buf][krow * KLD + kpart * 8]);
    }
    int row = tid >> 2, part = tid & 3;
    cp16(Aw + aoff + (size_t)tid * 8, &ash[buf][row * KLD + part * 8]);
    cp16(Utw + off + (size_t)(dvb * DVB_ + row) * C_ + part * 8,
         &usht[buf][row * KLD + part * 8]);
  };

  v8f Sacc[4];
#pragma unroll
  for (int t = 0; t < 4; ++t)
#pragma unroll
    for (int r = 0; r < 8; ++r) Sacc[t][r] = 0.f;

  for (int i = tid; i < (DVB_ * SLD) / 8; i += 128)
    ((uint4*)Sb)[i] = make_uint4(0u, 0u, 0u, 0u);

  // prologue: chunks 0 and 1 in flight
  stage(0, 0);
  stage(1, 1);

  for (int ch = 0; ch < NC_; ++ch) {
    const int cur = ch % 3;

    if (ch + 2 < NC_) {
      stage(ch + 2, (ch + 2) % 3);  // overwrites buffer retired at ch-1
      STAGE_WAIT(28);               // oldest in-flight (chunk ch) complete
    } else if (ch + 1 < NC_) {
      STAGE_WAIT(14);
    } else {
      STAGE_WAIT(0);
    }
    __syncthreads();

    const bf16* q_s = qsh[cur];
    const bf16* w_s = wsh[cur];
    const bf16* kt_s = ktsh[cur];
    const bf16* a_s = ash[cur];
    const bf16* u_s = usht[cur];

    // hoist state B fragments (shared by w@S and q@S)
    v16bf sbf[4];
#pragma unroll
    for (int kt = 0; kt < 4; ++kt)
      sbf[kt] = load_b_frag_col(Sb, SLD, kt * C_, ni * 16, lane);

    // interleaved independent chains: t1 = w@S, o = q@S
    v8f t1, o;
#pragma unroll
    for (int r = 0; r < 8; ++r) { t1[r] = 0.f; o[r] = 0.f; }
#pragma unroll
    for (int kt = 0; kt < 4; ++kt) {
      v16bf aw_ = load_a_frag(w_s, QLD, mi * 16, kt * C_, lane);
      t1 = wmma_bf16(aw_, sbf[kt], t1);
      v16bf aq_ = load_a_frag(q_s, QLD, mi * 16, kt * C_, lane);
      o = wmma_bf16(aq_, sbf[kt], o);
    }

    // u_s = u_pre - t1 -> ubt (column-major, one uint4 per lane)
    {
      int ncol = ni * 16 + nn;
      int mb   = mi * 16 + mhalf;
      uint4 u4 = *(const uint4*)(u_s + ncol * KLD + mb);
      uint4 s4;
      s4.x = pack2(bflo(u4.x) - t1[0], bfhi(u4.x) - t1[1]);
      s4.y = pack2(bflo(u4.y) - t1[2], bfhi(u4.y) - t1[3]);
      s4.z = pack2(bflo(u4.z) - t1[4], bfhi(u4.z) - t1[5]);
      s4.w = pack2(bflo(u4.w) - t1[6], bfhi(u4.w) - t1[7]);
      *(uint4*)(ubt + ncol * KLD + mb) = s4;
    }
    __syncthreads();

    // o += attn @ u_s ; write output tile
    {
      v16bf a = load_a_frag(a_s, KLD, mi * 16, 0, lane);
      v16bf b = load_b_frag_col(ubt, KLD, 0, ni * 16, lane);
      o = wmma_bf16(a, b, o);
    }
    {
      float* op = outg + ((size_t)bh * L_ + ch * C_) * D_ + dvb * DVB_;
      int ncol  = ni * 16 + nn;
      int mb    = mi * 16 + mhalf;
#pragma unroll
      for (int r = 0; r < 8; ++r)
        op[(size_t)(mb + r) * D_ + ncol] = o[r];
    }

    // S += k^T @ u_s (wave owns dk tiles 2wv, 2wv+1)
    {
      v16bf b0 = load_b_frag_col(ubt, KLD, 0, 0, lane);
      v16bf b1 = load_b_frag_col(ubt, KLD, 0, 16, lane);
#pragma unroll
      for (int m2 = 0; m2 < 2; ++m2) {
        v16bf a = load_a_frag(kt_s, KLD, (wv * 2 + m2) * 16, 0, lane);
        Sacc[m2 * 2 + 0] = wmma_bf16(a, b0, Sacc[m2 * 2 + 0]);
        Sacc[m2 * 2 + 1] = wmma_bf16(a, b1, Sacc[m2 * 2 + 1]);
      }
    }
    __syncthreads();   // all reads of Sb/ubt/stage[cur] complete

    // refresh bf16 state shadow
#pragma unroll
    for (int t = 0; t < 4; ++t) {
      int mb   = (wv * 2 + (t >> 1)) * 16 + mhalf;
      int ncol = (t & 1) * 16 + nn;
      uint4 s4;
      s4.x = pack2(Sacc[t][0], Sacc[t][1]);
      s4.y = pack2(Sacc[t][2], Sacc[t][3]);
      s4.z = pack2(Sacc[t][4], Sacc[t][5]);
      s4.w = pack2(Sacc[t][6], Sacc[t][7]);
      *(uint4*)(Sb + ncol * SLD + mb) = s4;
    }
  }

  // final state (f32) -> second output
  {
    float* sp = Sg + (size_t)bh * D_ * D_ + dvb * DVB_;
#pragma unroll
    for (int t = 0; t < 4; ++t) {
      int mb   = (wv * 2 + (t >> 1)) * 16 + mhalf;
      int ncol = (t & 1) * 16 + nn;
#pragma unroll
      for (int r = 0; r < 8; ++r)
        sp[(size_t)(mb + r) * D_ + ncol] = Sacc[t][r];
    }
  }
}

// ---------------------------------------------------------------------------
extern "C" void kernel_launch(void* const* d_in, const int* in_sizes, int n_in,
                              void* d_out, int out_size, void* d_ws,
                              size_t ws_size, hipStream_t stream) {
  (void)in_sizes; (void)n_in; (void)out_size; (void)ws_size;
  const float* q    = (const float*)d_in[0];
  const float* k    = (const float*)d_in[1];
  const float* v    = (const float*)d_in[2];
  const float* beta = (const float*)d_in[3];
  // d_in[4] = chunk_size (32, fixed at compile time)

  float* out  = (float*)d_out;
  float* Sout = out + (size_t)BH_ * L_ * D_;   // second tuple element

  // workspace (bf16): Q | Kt | W | Ut | attn  (~68 MB)
  bf16* Qw  = (bf16*)d_ws;
  bf16* Ktw = Qw  + (size_t)NCH_ * C_ * D_;
  bf16* Ww  = Ktw + (size_t)NCH_ * C_ * D_;
  bf16* Utw = Ww  + (size_t)NCH_ * C_ * D_;
  bf16* Aw  = Utw + (size_t)NCH_ * C_ * D_;

  deltanet_prep<<<NCH_, 128, 0, stream>>>(q, k, v, beta, Qw, Ktw, Ww, Utw, Aw);
  deltanet_scan<<<BH_ * NSL_, 128, 0, stream>>>(Qw, Ktw, Ww, Utw, Aw, out, Sout);
}
